// MultiHeadAttention_90211493085554
// MI455X (gfx1250) — matrix operand order repaired and checked by reading, not compiled, not run on hardware
//
#include <hip/hip_runtime.h>
#include <hip/hip_bf16.h>

// ---------------------------------------------------------------------------
// MI455X (gfx1250) MHA forward: bf16 WMMA everywhere (v_wmma_f32_16x16x32_bf16).
// Compute-bound problem (~69 GFLOP vs ~50MB of traffic -> ~2us at 23.3TB/s),
// so all four GEMMs and both attention matmuls go through the matrix pipe.
// wave32, one wave per workgroup, 16x64 f32-accumulator tile per wave.
// ---------------------------------------------------------------------------

typedef __bf16 bf16;
typedef __attribute__((ext_vector_type(16))) __bf16 v16bf;
typedef __attribute__((ext_vector_type(8)))  __bf16 v8bf;
typedef __attribute__((ext_vector_type(8)))  float  v8f;

#define D_MODEL 1024
#define N_HEADS 16
#define HEAD_DIM 64
#define SEQ 2048
#define BATCH 2
#define NROWS (BATCH * SEQ) /* 4096 */

__device__ __forceinline__ bf16 f2bf(float f) {
  unsigned u = __builtin_bit_cast(unsigned, f);
  u += 0x7FFFu + ((u >> 16) & 1u);            // round-to-nearest-even
  unsigned short s = (unsigned short)(u >> 16);
  return __builtin_bit_cast(bf16, s);
}

// 16 bf16 halves from two contiguous 16B chunks (lowers to global_load_b128 x2)
__device__ __forceinline__ v16bf ld16(const bf16* p0, const bf16* p1) {
  v8bf x = *(const v8bf*)p0;
  v8bf y = *(const v8bf*)p1;
  v16bf a;
#pragma unroll
  for (int i = 0; i < 8; ++i) { a[i] = x[i]; a[i + 8] = y[i]; }
  return a;
}

__device__ __forceinline__ v8f wmma_bf(v16bf a, v16bf b, v8f c) {
  // (neg_a, A, neg_b, B, c_mod, C, reuse_a, reuse_b)
  return __builtin_amdgcn_wmma_f32_16x16x32_bf16(false, a, false, b, (short)0, c,
                                                 false, false);
}

// ---------------------------------------------------------------------------
// Kernel 1: fp32 -> bf16 conversion (x and weights), 4 elements/thread
// ---------------------------------------------------------------------------
__global__ void cvt_bf16(const float* __restrict__ s, bf16* __restrict__ d, int n) {
  int i = (blockIdx.x * blockDim.x + threadIdx.x) * 4;
  if (i < n) {
    float4 v = *(const float4*)(s + i);
    d[i + 0] = f2bf(v.x);
    d[i + 1] = f2bf(v.y);
    d[i + 2] = f2bf(v.z);
    d[i + 3] = f2bf(v.w);
  }
}

// ---------------------------------------------------------------------------
// Kernel 2: fused Q/K/V projection + RoPE.
//   grid = (NROWS/16, 3*N_HEADS), block = 32 (one wave).
//   blockIdx.y: [0,16) -> Q head h, [16,32) -> K head, [32,48) -> V head.
//   Q,K: RoPE'd, stored (b,h,s,d).  V: stored transposed (b,h,d,s).
// ---------------------------------------------------------------------------
__global__ void __launch_bounds__(32)
qkv_rope(const bf16* __restrict__ X, const bf16* __restrict__ Wq,
         const bf16* __restrict__ Wk, const bf16* __restrict__ Wv,
         bf16* __restrict__ Qr, bf16* __restrict__ Kr, bf16* __restrict__ Vt) {
  __shared__ float lds[16][64];
  const int l  = threadIdx.x;
  const int lm = l & 15, lg = l >> 4;
  const int mt = blockIdx.x;      // 16-row tile of (B*S)
  const int w  = blockIdx.y >> 4; // 0=Q 1=K 2=V
  const int h  = blockIdx.y & 15; // head == 64-col tile of D_MODEL

  const bf16* W = (w == 0) ? Wq : (w == 1 ? Wk : Wv);
  const bf16* xrow = X + (size_t)(mt * 16 + lm) * D_MODEL;

  v8f acc[4] = {};
  for (int k0 = 0; k0 < D_MODEL; k0 += 32) {
    // A: x tile 16x32 (this code's k order, same as B: halves 0..15 -> k=16*lg+j)
    v16bf a = ld16(xrow + k0 + 16 * lg, xrow + k0 + 16 * lg + 8);
#pragma unroll
    for (int t = 0; t < 4; ++t) {
      // B: W^T tile 32x16; lane holds out-col n = h*64+16t+lm, halves k=16*lg+j
      const bf16* wr = W + (size_t)(h * 64 + 16 * t + lm) * D_MODEL + k0 + 16 * lg;
      v16bf b = ld16(wr, wr + 8);
      acc[t] = wmma_bf(a, b, acc[t]);
    }
  }

  const int bb    = (mt * 16) >> 11;   // batch
  const int sbase = (mt * 16) & (SEQ - 1);

  if (w == 2) {
    // V transposed store: Vt[(b*H+h)*64 + c][s]; lane owns col c, 8 contiguous s
    bf16* vb = Vt + (size_t)(bb * N_HEADS + h) * HEAD_DIM * SEQ;
#pragma unroll
    for (int t = 0; t < 4; ++t) {
      int c = 16 * t + lm;
      v8bf o;
#pragma unroll
      for (int r = 0; r < 8; ++r) o[r] = f2bf(acc[t][r]);
      *(v8bf*)(vb + (size_t)c * SEQ + sbase + 8 * lg) = o;
    }
  } else {
    // stage f32 tile in LDS, then apply RoPE (pairs (2i,2i+1) -> (i, i+32))
#pragma unroll
    for (int t = 0; t < 4; ++t)
#pragma unroll
      for (int r = 0; r < 8; ++r)
        lds[r + 8 * lg][16 * t + lm] = acc[t][r];
    __syncthreads();

    bf16* dst = ((w == 0) ? Qr : Kr) + (size_t)(bb * N_HEADS + h) * SEQ * HEAD_DIM;
    const int i = l;                         // freq index 0..31
    // inv_freq = 10000^(-2i/64) = exp(-i * ln(1e4)/32)
    float invf = __expf(-0.28782313662425572f * (float)i);
#pragma unroll 4
    for (int m = 0; m < 16; ++m) {
      float pos = (float)(sbase + m);
      float sn, cs;
      __sincosf(pos * invf, &sn, &cs);
      float t1 = lds[m][2 * i], t2 = lds[m][2 * i + 1];
      bf16* o = dst + (size_t)(sbase + m) * HEAD_DIM;
      o[i]      = f2bf(t1 * cs - t2 * sn);
      o[i + 32] = f2bf(t1 * sn + t2 * cs);
    }
  }
}

// ---------------------------------------------------------------------------
// Kernel 3: causal flash attention, one wave per 16-row Q tile.
//   grid = (SEQ/16, BATCH*N_HEADS), block = 32.
//   KV processed in 32-row blocks: 2x score WMMAs (2 fragments each), online
//   softmax via shfl_xor over 16-lane groups, P->A fragment via LDS, 4x PV WMMAs.
// ---------------------------------------------------------------------------
__global__ void __launch_bounds__(32)
attn(const bf16* __restrict__ Qr, const bf16* __restrict__ Kr,
     const bf16* __restrict__ Vt, bf16* __restrict__ Ob) {
  __shared__ float lp[16][32];
  const int l  = threadIdx.x;
  const int lm = l & 15, lg = l >> 4;
  const int qt = blockIdx.x;
  const int bh = blockIdx.y;
  const int bb = bh >> 4, h = bh & 15;

  const bf16* Qb = Qr + (size_t)bh * SEQ * HEAD_DIM;
  const bf16* Kb = Kr + (size_t)bh * SEQ * HEAD_DIM;
  const bf16* Vb = Vt + (size_t)bh * HEAD_DIM * SEQ;

  // Q tile as two A fragments (c 0..31 and 32..63)
  const bf16* qrow = Qb + (size_t)(qt * 16 + lm) * HEAD_DIM;
  v16bf aq0 = ld16(qrow + 16 * lg,      qrow + 16 * lg + 8);
  v16bf aq1 = ld16(qrow + 32 + 16 * lg, qrow + 40 + 16 * lg);

  v8f oacc[4] = {};
  float rm[8], rl[8];
#pragma unroll
  for (int r = 0; r < 8; ++r) { rm[r] = -1e30f; rl[r] = 0.0f; }

  const int jbmax = (qt * 16 + 15) / 32;
  for (int jb = 0; jb <= jbmax; ++jb) {
    const int kv0 = jb * 32;

    // scores S = Q K^T for kv rows [kv0, kv0+32): two 16x16 tiles
    v8f sa[2];
#pragma unroll
    for (int u = 0; u < 2; ++u) {
      const bf16* krow = Kb + (size_t)(kv0 + 16 * u + lm) * HEAD_DIM;
      v16bf bk0 = ld16(krow + 16 * lg,      krow + 16 * lg + 8);       // c 0..31
      v16bf bk1 = ld16(krow + 32 + 16 * lg, krow + 40 + 16 * lg);      // c 32..63
      v8f s = {};
      s = wmma_bf(aq0, bk0, s);
      s = wmma_bf(aq1, bk1, s);
      sa[u] = s;
    }

    // scale, causal mask, online softmax statistics
    float al[8], p0[8], p1[8];
#pragma unroll
    for (int r = 0; r < 8; ++r) {
      int qi = qt * 16 + r + 8 * lg;
      float s0 = sa[0][r] * 0.125f;          // 1/sqrt(64)
      float s1 = sa[1][r] * 0.125f;
      if (kv0 + lm > qi)      s0 = -1e30f;
      if (kv0 + 16 + lm > qi) s1 = -1e30f;
      float mx = fmaxf(s0, s1);
#pragma unroll
      for (int msk = 1; msk <= 8; msk <<= 1)
        mx = fmaxf(mx, __shfl_xor(mx, msk, 32));
      float mnew = fmaxf(rm[r], mx);
      float a  = __expf(rm[r] - mnew);
      float e0 = __expf(s0 - mnew);
      float e1 = __expf(s1 - mnew);
      float ps = e0 + e1;
#pragma unroll
      for (int msk = 1; msk <= 8; msk <<= 1)
        ps += __shfl_xor(ps, msk, 32);
      rl[r] = rl[r] * a + ps;
      rm[r] = mnew;
      al[r] = a;
      p0[r] = e0;
      p1[r] = e1;
    }

    // P (C-layout) -> LDS -> A-fragment layout
#pragma unroll
    for (int r = 0; r < 8; ++r) {
      lp[r + 8 * lg][lm]      = p0[r];
      lp[r + 8 * lg][16 + lm] = p1[r];
    }
    __syncthreads();

#pragma unroll
    for (int t = 0; t < 4; ++t)
#pragma unroll
      for (int r = 0; r < 8; ++r)
        oacc[t][r] *= al[r];

    v16bf ap;
#pragma unroll
    for (int j = 0; j < 16; ++j) {
      int k = j + 16 * lg;   // contiguous k order, same as the V B-fragment
      ap[j] = f2bf(lp[lm][k]);
    }
    __syncthreads();

    // O += P * V : B fragments from transposed V are contiguous 32B loads
#pragma unroll
    for (int t = 0; t < 4; ++t) {
      const bf16* vcol = Vb + (size_t)(16 * t + lm) * SEQ + kv0 + 16 * lg;
      v16bf bv = ld16(vcol, vcol + 8);
      oacc[t] = wmma_bf(ap, bv, oacc[t]);
    }
  }

  // normalize and store O as bf16 in (b, s, h*64+c) layout for the out-proj
#pragma unroll
  for (int t = 0; t < 4; ++t) {
    int c = h * HEAD_DIM + 16 * t + lm;
#pragma unroll
    for (int r = 0; r < 8; ++r) {
      int s = qt * 16 + r + 8 * lg;
      Ob[(size_t)(bb * SEQ + s) * D_MODEL + c] = f2bf(oacc[t][r] / rl[r]);
    }
  }
}

// ---------------------------------------------------------------------------
// Kernel 4: out = O @ Wo^T, fp32 result straight to d_out.
//   grid = (NROWS/16, D_MODEL/64), block = 32.
// ---------------------------------------------------------------------------
__global__ void __launch_bounds__(32)
oproj(const bf16* __restrict__ Ob, const bf16* __restrict__ Wo,
      float* __restrict__ out) {
  const int l  = threadIdx.x;
  const int lm = l & 15, lg = l >> 4;
  const int mt = blockIdx.x, nt = blockIdx.y;

  v8f acc[4] = {};
  const bf16* xrow = Ob + (size_t)(mt * 16 + lm) * D_MODEL;
  for (int k0 = 0; k0 < D_MODEL; k0 += 32) {
    v16bf a = ld16(xrow + k0 + 16 * lg, xrow + k0 + 16 * lg + 8);
#pragma unroll
    for (int t = 0; t < 4; ++t) {
      const bf16* wr = Wo + (size_t)(nt * 64 + 16 * t + lm) * D_MODEL + k0 + 16 * lg;
      v16bf b = ld16(wr, wr + 8);
      acc[t] = wmma_bf(a, b, acc[t]);
    }
  }
#pragma unroll
  for (int t = 0; t < 4; ++t) {
    int n = nt * 64 + 16 * t + lm;
#pragma unroll
    for (int r = 0; r < 8; ++r)
      out[(size_t)(mt * 16 + r + 8 * lg) * D_MODEL + n] = acc[t][r];
  }
}

// ---------------------------------------------------------------------------
extern "C" void kernel_launch(void* const* d_in, const int* in_sizes, int n_in,
                              void* d_out, int out_size, void* d_ws, size_t ws_size,
                              hipStream_t stream) {
  const float* x  = (const float*)d_in[0];
  const float* Wq = (const float*)d_in[1];
  const float* Wk = (const float*)d_in[2];
  const float* Wv = (const float*)d_in[3];
  const float* Wo = (const float*)d_in[4];
  float* out = (float*)d_out;

  char* ws = (char*)d_ws;
  const size_t MB = 1u << 20;
  bf16* xb  = (bf16*)(ws + 0 * MB);   // 4096x1024 bf16  (8 MB)
  bf16* wqb = (bf16*)(ws + 8 * MB);   // 1024x1024 bf16  (2 MB each)
  bf16* wkb = (bf16*)(ws + 10 * MB);
  bf16* wvb = (bf16*)(ws + 12 * MB);
  bf16* wob = (bf16*)(ws + 14 * MB);
  bf16* Qr  = (bf16*)(ws + 16 * MB);  // (b,h,s,d) bf16  (8 MB)
  bf16* Kr  = (bf16*)(ws + 24 * MB);  // (b,h,s,d)
  bf16* Vt  = (bf16*)(ws + 32 * MB);  // (b,h,d,s) transposed
  bf16* Ob  = (bf16*)(ws + 40 * MB);  // (b,s,D) attention output

  const int nx = NROWS * D_MODEL;     // 4M elems
  const int nw = D_MODEL * D_MODEL;   // 1M elems
  cvt_bf16<<<(nx / 4 + 255) / 256, 256, 0, stream>>>(x,  xb,  nx);
  cvt_bf16<<<(nw / 4 + 255) / 256, 256, 0, stream>>>(Wq, wqb, nw);
  cvt_bf16<<<(nw / 4 + 255) / 256, 256, 0, stream>>>(Wk, wkb, nw);
  cvt_bf16<<<(nw / 4 + 255) / 256, 256, 0, stream>>>(Wv, wvb, nw);
  cvt_bf16<<<(nw / 4 + 255) / 256, 256, 0, stream>>>(Wo, wob, nw);

  qkv_rope<<<dim3(NROWS / 16, 3 * N_HEADS), 32, 0, stream>>>(xb, wqb, wkb, wvb,
                                                             Qr, Kr, Vt);
  attn<<<dim3(SEQ / 16, BATCH * N_HEADS), 32, 0, stream>>>(Qr, Kr, Vt, Ob);
  oproj<<<dim3(NROWS / 16, D_MODEL / 64), 32, 0, stream>>>(Ob, wob, out);
}
